// decoder_57251914055985
// MI455X (gfx1250) — matrix operand order repaired and checked by reading, not compiled, not run on hardware
//
#include <hip/hip_runtime.h>
#include <hip/hip_bf16.h>
#include <math.h>

// ---------------- problem constants ----------------
constexpr int BB    = 32;    // batch
constexpr int TENC  = 1024;  // encoder timesteps
constexpr int H2    = 1024;  // hidden
constexpr int EE    = 128;   // embed / vocab
constexpr int LL    = 128;   // decode steps
constexpr int GG    = 4096;  // 4*H2 gates
constexpr int IN    = 1152;  // E + H2
constexpr int MX    = LL * BB;   // 4096 rows of xs / preds
constexpr int KOUT  = 2048;      // 2*H2

typedef __bf16 bf16_t;
typedef __attribute__((ext_vector_type(16))) __bf16 v16bf;
typedef __attribute__((ext_vector_type(8)))  __bf16 v8bf;
typedef __attribute__((ext_vector_type(8)))  float  v8f;
typedef __attribute__((ext_vector_type(4)))  unsigned int v4u;
typedef __attribute__((ext_vector_type(8)))  int v8i_;
typedef __attribute__((ext_vector_type(4)))  int v4i_;

// ---------------- WMMA helpers (CDNA5 wave32, 16x16x32 bf16) ----------------
__device__ __forceinline__ v8f wmma_bf16(v16bf a, v16bf b, v8f c) {
  return __builtin_amdgcn_wmma_f32_16x16x32_bf16(
      false, a, false, b, (short)0, c, false, false);
}

// A-matrix 16x32 bf16 fragment. Lane (half=lane>>4, m=lane&15) holds
// frag[0..7]  = row m, k = half*16 + 0..7       (contiguous)
// frag[8..15] = row m, k = half*16 + 8 + 0..7   (contiguous)
__device__ __forceinline__ v16bf load_frag_a(const bf16_t* __restrict__ base,
                                             int row_stride, int half, int m) {
  const bf16_t* rp = base + (size_t)m * row_stride;
  v8bf lo = *(const v8bf*)(rp + half * 16);
  v8bf hi = *(const v8bf*)(rp + half * 16 + 8);
  v16bf a;
#pragma unroll
  for (int i = 0; i < 8; ++i) { a[i] = lo[i]; a[8 + i] = hi[i]; }
  return a;
}

// B-matrix 32x16 fragment from an n-major source (row n, contiguous k).
// Caller passes p = &src[n_row][k0 + half*16]. Works for global or LDS source.
__device__ __forceinline__ v16bf load_frag_b_nk(const bf16_t* p) {
  v8bf lo = *(const v8bf*)(p);
  v8bf hi = *(const v8bf*)(p + 8);
  v16bf b;
#pragma unroll
  for (int i = 0; i < 8; ++i) { b[i] = lo[i]; b[8 + i] = hi[i]; }
  return b;
}

// B-matrix 32x16 fragment from a k-major source (src[k][n], row stride given).
__device__ __forceinline__ v16bf load_frag_b_kn(const bf16_t* __restrict__ base,
                                                int row_stride, int half, int n) {
  const bf16_t* p = base + (size_t)(half * 16) * row_stride + n;
  v16bf b;
#pragma unroll
  for (int e = 0; e < 16; ++e) b[e] = p[(size_t)e * row_stride];
  return b;
}

__device__ __forceinline__ float sigm(float x) { return 1.f / (1.f + expf(-x)); }

// ---------------- prep: f32 -> bf16 conversions ----------------
__global__ __launch_bounds__(256) void cvt_bf16(const float* __restrict__ src,
                                                bf16_t* __restrict__ dst, long n) {
  long i = (long)blockIdx.x * blockDim.x + threadIdx.x;
  long st = (long)gridDim.x * blockDim.x;
  for (; i < n; i += st) dst[i] = (bf16_t)src[i];
}

// xs[m][k], m = t*32 + b, k<EE -> teacher-forced word, else ctx0[b][k-EE]
__global__ __launch_bounds__(256) void build_xs(const float* __restrict__ gt,
                                                const float* __restrict__ ctx,
                                                bf16_t* __restrict__ xs) {
  long total = (long)MX * IN;
  long i = (long)blockIdx.x * blockDim.x + threadIdx.x;
  long st = (long)gridDim.x * blockDim.x;
  for (; i < total; i += st) {
    int m = (int)(i / IN), k = (int)(i % IN);
    int t = m >> 5, b = m & 31;
    float v;
    if (k < EE)
      v = (t == 0) ? (k == 0 ? 1.f : 0.f)
                   : gt[((size_t)b * LL + (t - 1)) * EE + k];
    else
      v = ctx[(size_t)b * TENC * H2 + (k - EE)];  // context[b][0][k-EE]
    xs[i] = (bf16_t)v;
  }
}

// ---------------- kernel A: igates = xs @ W_ih^T + (b_ih + b_hh) ----------------
__global__ __launch_bounds__(256) void igates_gemm(
    const bf16_t* __restrict__ xs, const bf16_t* __restrict__ Wih,
    const float* __restrict__ bih, const float* __restrict__ bhh,
    float* __restrict__ igates) {
  int wave = threadIdx.x >> 5, lane = threadIdx.x & 31;
  int half = lane >> 4, ln = lane & 15;
  int n0 = blockIdx.x * 128 + wave * 16;
  int m0 = blockIdx.y * 32;
  const bf16_t* arow = xs + (size_t)m0 * IN;
  const bf16_t* brow = Wih + (size_t)(n0 + ln) * IN;
  v8f acc0 = {}, acc1 = {};
  for (int kc = 0; kc < IN; kc += 32) {
    v16bf a0 = load_frag_a(arow + kc, IN, half, ln);
    v16bf a1 = load_frag_a(arow + (size_t)16 * IN + kc, IN, half, ln);
    v16bf b  = load_frag_b_nk(brow + kc + half * 16);
    acc0 = wmma_bf16(a0, b, acc0);
    acc1 = wmma_bf16(a1, b, acc1);
  }
  int nc = n0 + ln;
  float bias = bih[nc] + bhh[nc];
#pragma unroll
  for (int r = 0; r < 8; ++r) {
    int m = m0 + r + half * 8;
    igates[(size_t)m * GG + nc]        = acc0[r] + bias;
    igates[(size_t)(m + 16) * GG + nc] = acc1[r] + bias;
  }
}

// ---------------- kernel B: persistent LSTM recurrence ----------------
// 64 WGs x 8 waves. WG owns h/c columns [j0, j0+16). Wave (gate = w&3,
// kh = w>>2) computes a 32x16 K-half partial of (h_{t-1} @ W_hh^T).
// W_hh slice (64 rows x 1024, 128 KB bf16) is TDM-staged into LDS once and
// reused for all 128 steps. Cell state c lives in LDS. One device-wide
// barrier per step (monotonic counter in ws, memset per launch).
__global__ __launch_bounds__(256) void lstm_recur(
    const float* __restrict__ igates, const bf16_t* __restrict__ Whh,
    bf16_t* __restrict__ h_all, unsigned int* __restrict__ bar) {
  __shared__ bf16_t whh_lds[64 * H2];    // 128 KB: row lg = gate*16 + j
  __shared__ float pgates[2 * 32 * 64];  // 16 KB: [kh][batch][gate*16 + j]
  __shared__ float c_lds[32 * 16];       // 2 KB:  [batch][j]
  const int j0 = blockIdx.x * 16;
  const int wave = threadIdx.x >> 5;
  const int gate = wave & 3, kh = wave >> 2;
  const int lane = threadIdx.x & 31;
  const int half = lane >> 4, ln = lane & 15;

  for (int i = threadIdx.x; i < 32 * 16; i += 256) c_lds[i] = 0.f;
  for (int i = threadIdx.x; i < 2 * 32 * 64; i += 256) pgates[i] = 0.f;

  // ---- stage W_hh rows {g*1024 + j0 .. +16} for g=0..3 into LDS ----
#if __has_builtin(__builtin_amdgcn_tensor_load_to_lds)
  if (wave == 0) {
    unsigned lds_base = (unsigned)(unsigned long long)(void*)whh_lds;
#pragma unroll
    for (int g = 0; g < 4; ++g) {
      unsigned long long ga =
          (unsigned long long)(size_t)(const void*)(Whh +
              ((size_t)g * 1024 + j0) * H2);
      v4u g0; v8i_ g1;
      v4i_ g2 = {0, 0, 0, 0}, g3 = {0, 0, 0, 0};
      v8i_ g4 = {0, 0, 0, 0, 0, 0, 0, 0};
      g0[0] = 1u;                                      // count=1, normal mode
      g0[1] = lds_base + (unsigned)(g * 16 * H2 * 2);  // lds_addr (bytes)
      g0[2] = (unsigned)(ga & 0xffffffffu);            // global_addr[31:0]
      g0[3] = (unsigned)((ga >> 32) & 0x01ffffffu)     // global_addr[56:32]
              | 0x80000000u;                           // type=2 ("image")
      g1[0] = 0x00010000;   // data_size=1 (2 bytes/elem)
      g1[1] = 0x04000000;   // tensor_dim0 = 1024  (bits 79:48)
      g1[2] = 0x10000000;   // tensor_dim1 = 4096  (bits 111:80)
      g1[3] = 0x04000000;   // tile_dim0   = 1024  (bits 127:112)
      g1[4] = 16;           // tile_dim1   = 16
      g1[5] = (int)H2;      // tensor_dim0_stride = 1024 (bits 207:160)
      g1[6] = 0;
      g1[7] = 0;
      __builtin_amdgcn_tensor_load_to_lds(g0, g1, g2, g3, g4, 0);
    }
  }
  __builtin_amdgcn_s_wait_tensorcnt(0);
#else
  for (int i = threadIdx.x; i < 64 * (H2 / 8); i += 256) {
    int lg = i >> 7;            // row 0..63
    int ko = (i & 127) * 8;     // 8 bf16 chunks
    int row = (lg >> 4) * 1024 + j0 + (lg & 15);
    *(v8bf*)&whh_lds[lg * H2 + ko] = *(const v8bf*)(Whh + (size_t)row * H2 + ko);
  }
#endif
  __syncthreads();

  const bf16_t* wrow_lds = whh_lds + (size_t)(gate * 16 + ln) * H2;

  for (int t = 0; t < LL; ++t) {
    v8f acc0 = {}, acc1 = {};
    if (t > 0) {
      const bf16_t* hprev = h_all + (size_t)(t - 1) * BB * H2;
      const int kbeg = kh * 512, kend = kbeg + 512;
      for (int kc = kbeg; kc < kend; kc += 32) {
        v16bf a0 = load_frag_a(hprev + kc, H2, half, ln);
        v16bf a1 = load_frag_a(hprev + (size_t)16 * H2 + kc, H2, half, ln);
        v16bf b  = load_frag_b_nk(wrow_lds + kc + half * 16);
        acc0 = wmma_bf16(a0, b, acc0);
        acc1 = wmma_bf16(a1, b, acc1);
      }
    }
    float* pg = pgates + kh * 32 * 64;
#pragma unroll
    for (int r = 0; r < 8; ++r) {
      pg[(r + half * 8) * 64 + gate * 16 + ln]      = acc0[r];
      pg[(16 + r + half * 8) * 64 + gate * 16 + ln] = acc1[r];
    }
    __syncthreads();
    const float* ig = igates + (size_t)t * BB * GG;
    for (int idx = threadIdx.x; idx < 32 * 16; idx += 256) {
      int m = idx >> 4, j = idx & 15;
      const float* p0 = &pgates[m * 64 + j];
      const float* p1 = &pgates[32 * 64 + m * 64 + j];
      const float* igr = ig + (size_t)m * GG + j0 + j;
      float iv = sigm(p0[0] + p1[0] + igr[0]);
      float fv = sigm(p0[16] + p1[16] + igr[1024]);
      float gv = tanhf(p0[32] + p1[32] + igr[2048]);
      float ov = sigm(p0[48] + p1[48] + igr[3072]);
      float cn = fv * c_lds[idx] + iv * gv;
      c_lds[idx] = cn;
      h_all[(size_t)t * BB * H2 + (size_t)m * H2 + j0 + j] = (bf16_t)(ov * tanhf(cn));
      if (t + 1 < LL) __builtin_prefetch(igr + (size_t)BB * GG, 0, 1);
    }
    __syncthreads();
    __threadfence();
    if (threadIdx.x == 0) {
      atomicAdd(bar, 1u);
      unsigned tgt = (unsigned)(t + 1) * gridDim.x;
      while (atomicAdd(bar, 0u) < tgt) { __builtin_amdgcn_s_sleep(8); }
    }
    __syncthreads();
    __threadfence();
  }
}

// ---------------- kernel C1: energy[b] = H_b (LxH2) @ context_b^T ----------------
__global__ __launch_bounds__(256) void attn_energy(
    const bf16_t* __restrict__ h_all, const bf16_t* __restrict__ ctxb,
    float* __restrict__ energy) {
  int sub = blockIdx.x & 31, b = blockIdx.x >> 5;
  int mb = sub >> 3, nb = sub & 7;
  int wave = threadIdx.x >> 5, lane = threadIdx.x & 31;
  int half = lane >> 4, ln = lane & 15;
  int m0 = mb * 32, n0 = nb * 128 + wave * 16;
  const bf16_t* abase = h_all + (size_t)m0 * BB * H2 + (size_t)b * H2;
  const bf16_t* brow  = ctxb + (size_t)b * TENC * H2 + (size_t)(n0 + ln) * H2;
  v8f acc0 = {}, acc1 = {};
  for (int kc = 0; kc < H2; kc += 32) {
    v16bf a0 = load_frag_a(abase + kc, BB * H2, half, ln);
    v16bf a1 = load_frag_a(abase + (size_t)16 * BB * H2 + kc, BB * H2, half, ln);
    v16bf bf = load_frag_b_nk(brow + kc + half * 16);
    acc0 = wmma_bf16(a0, bf, acc0);
    acc1 = wmma_bf16(a1, bf, acc1);
  }
  float* eb = energy + (size_t)b * LL * TENC;
  int nc = n0 + ln;
#pragma unroll
  for (int r = 0; r < 8; ++r) {
    eb[(size_t)(m0 + r + half * 8) * TENC + nc]      = acc0[r];
    eb[(size_t)(m0 + 16 + r + half * 8) * TENC + nc] = acc1[r];
  }
}

// ---------------- kernel C2: row softmax (len 1024) -> bf16 scores ----------------
__global__ __launch_bounds__(256) void softmax_rows(const float* __restrict__ energy,
                                                    bf16_t* __restrict__ score) {
  int row = blockIdx.x;  // b*LL + t_dec
  const float* e = energy + (size_t)row * TENC;
  __shared__ float red[256];
  float mx = -INFINITY;
  for (int i = threadIdx.x; i < TENC; i += 256) mx = fmaxf(mx, e[i]);
  red[threadIdx.x] = mx; __syncthreads();
  for (int s = 128; s > 0; s >>= 1) {
    if (threadIdx.x < s) red[threadIdx.x] = fmaxf(red[threadIdx.x], red[threadIdx.x + s]);
    __syncthreads();
  }
  mx = red[0]; __syncthreads();
  float sum = 0.f;
  for (int i = threadIdx.x; i < TENC; i += 256) sum += expf(e[i] - mx);
  red[threadIdx.x] = sum; __syncthreads();
  for (int s = 128; s > 0; s >>= 1) {
    if (threadIdx.x < s) red[threadIdx.x] += red[threadIdx.x + s];
    __syncthreads();
  }
  float inv = 1.f / red[0];
  bf16_t* sr = score + (size_t)row * TENC;
  for (int i = threadIdx.x; i < TENC; i += 256) sr[i] = (bf16_t)(expf(e[i] - mx) * inv);
}

// ---------------- kernel C3: ctx[b] = score_b (LxT) @ context_b (TxH2) ----------------
__global__ __launch_bounds__(256) void attn_ctx(
    const bf16_t* __restrict__ score, const bf16_t* __restrict__ ctxb,
    bf16_t* __restrict__ ctx_all) {
  int sub = blockIdx.x & 31, b = blockIdx.x >> 5;
  int mb = sub >> 3, nb = sub & 7;
  int wave = threadIdx.x >> 5, lane = threadIdx.x & 31;
  int half = lane >> 4, ln = lane & 15;
  int m0 = mb * 32, n0 = nb * 128 + wave * 16;
  const bf16_t* abase = score + (size_t)(b * LL + m0) * TENC;
  const bf16_t* bbase = ctxb + (size_t)b * TENC * H2 + n0;
  v8f acc0 = {}, acc1 = {};
  for (int kc = 0; kc < TENC; kc += 32) {
    v16bf a0 = load_frag_a(abase + kc, TENC, half, ln);
    v16bf a1 = load_frag_a(abase + (size_t)16 * TENC + kc, TENC, half, ln);
    v16bf bf = load_frag_b_kn(bbase + (size_t)kc * H2, H2, half, ln);
    acc0 = wmma_bf16(a0, bf, acc0);
    acc1 = wmma_bf16(a1, bf, acc1);
  }
  int nc = n0 + ln;
#pragma unroll
  for (int r = 0; r < 8; ++r) {
    ctx_all[(size_t)(m0 + r + half * 8) * BB * H2 + (size_t)b * H2 + nc] =
        (bf16_t)acc0[r];
    ctx_all[(size_t)(m0 + 16 + r + half * 8) * BB * H2 + (size_t)b * H2 + nc] =
        (bf16_t)acc1[r];
  }
}

// ---------------- kernel C4: logits = [h,ctx] @ W_out^T + b_out ----------------
__global__ __launch_bounds__(256) void out_logits(
    const bf16_t* __restrict__ h_all, const bf16_t* __restrict__ ctx_all,
    const bf16_t* __restrict__ Wout, const float* __restrict__ bout,
    float* __restrict__ logits) {
  int wave = threadIdx.x >> 5, lane = threadIdx.x & 31;
  int half = lane >> 4, ln = lane & 15;
  int m0 = blockIdx.x * 32;
  int n0 = wave * 16;
  const bf16_t* brow = Wout + (size_t)(n0 + ln) * KOUT;
  v8f acc0 = {}, acc1 = {};
  for (int kc = 0; kc < KOUT; kc += 32) {
    const bf16_t* ab = (kc < H2) ? (h_all + kc) : (ctx_all + (kc - H2));
    v16bf a0 = load_frag_a(ab + (size_t)m0 * H2, H2, half, ln);
    v16bf a1 = load_frag_a(ab + (size_t)(m0 + 16) * H2, H2, half, ln);
    v16bf bf = load_frag_b_nk(brow + kc + half * 16);
    acc0 = wmma_bf16(a0, bf, acc0);
    acc1 = wmma_bf16(a1, bf, acc1);
  }
  int nc = n0 + ln;
  float bias = bout[nc];
#pragma unroll
  for (int r = 0; r < 8; ++r) {
    logits[(size_t)(m0 + r + half * 8) * EE + nc]      = acc0[r] + bias;
    logits[(size_t)(m0 + 16 + r + half * 8) * EE + nc] = acc1[r] + bias;
  }
}

// ---------------- kernel C5: log_softmax over EE=128, write out[b][t][e] ----------------
__global__ __launch_bounds__(128) void out_logsoftmax(const float* __restrict__ logits,
                                                      float* __restrict__ out) {
  int m = blockIdx.x;  // t*32 + b
  __shared__ float red[128];
  float v = logits[(size_t)m * EE + threadIdx.x];
  red[threadIdx.x] = v; __syncthreads();
  for (int s = 64; s > 0; s >>= 1) {
    if (threadIdx.x < s) red[threadIdx.x] = fmaxf(red[threadIdx.x], red[threadIdx.x + s]);
    __syncthreads();
  }
  float mx = red[0]; __syncthreads();
  red[threadIdx.x] = expf(v - mx); __syncthreads();
  for (int s = 64; s > 0; s >>= 1) {
    if (threadIdx.x < s) red[threadIdx.x] += red[threadIdx.x + s];
    __syncthreads();
  }
  float lse = mx + logf(red[0]);
  int t = m >> 5, b = m & 31;
  out[(size_t)b * LL * EE + (size_t)t * EE + threadIdx.x] = v - lse;
}

// ---------------- launch ----------------
extern "C" void kernel_launch(void* const* d_in, const int* in_sizes, int n_in,
                              void* d_out, int out_size, void* d_ws, size_t ws_size,
                              hipStream_t stream) {
  const float* ctx_f  = (const float*)d_in[0];
  const float* gt     = (const float*)d_in[1];
  const float* Wih_f  = (const float*)d_in[2];
  const float* Whh_f  = (const float*)d_in[3];
  const float* bih    = (const float*)d_in[4];
  const float* bhh    = (const float*)d_in[5];
  const float* Wout_f = (const float*)d_in[6];
  const float* bout   = (const float*)d_in[7];
  float* out = (float*)d_out;
  char* ws = (char*)d_ws;

  // workspace layout (bytes)
  size_t o = 0;
  unsigned int* bar = (unsigned int*)(ws + o);              o += 256;
  bf16_t* xs_b   = (bf16_t*)(ws + o);  o += (size_t)MX * IN * 2;        // 9.4 MB
  bf16_t* wih_b  = (bf16_t*)(ws + o);  o += (size_t)GG * IN * 2;        // 9.4 MB
  bf16_t* whh_b  = (bf16_t*)(ws + o);  o += (size_t)GG * H2 * 2;        // 8.4 MB
  bf16_t* ctx_b  = (bf16_t*)(ws + o);  o += (size_t)BB * TENC * H2 * 2; // 67 MB
  bf16_t* wout_b = (bf16_t*)(ws + o);  o += (size_t)EE * KOUT * 2;      // 0.5 MB
  bf16_t* h_all  = (bf16_t*)(ws + o);  o += (size_t)LL * BB * H2 * 2;   // 8.4 MB
  char* big = ws + o;  // 64 MiB region, reused across phases
  float*  igates  = (float*)big;                       // phase 1/2 (64 MB)
  float*  energy  = (float*)big;                       // phase 3 (16 MB)
  bf16_t* score   = (bf16_t*)(big + 16777216);         // 8 MB
  bf16_t* ctx_all = (bf16_t*)(big + 25165824);         // 8 MB
  float*  logits  = (float*)(big + 33554432);          // 2 MB

  (void)hipMemsetAsync(bar, 0, 256, stream);

  cvt_bf16<<<2048, 256, 0, stream>>>(Wih_f, wih_b, (long)GG * IN);
  cvt_bf16<<<2048, 256, 0, stream>>>(Whh_f, whh_b, (long)GG * H2);
  cvt_bf16<<<4096, 256, 0, stream>>>(ctx_f, ctx_b, (long)BB * TENC * H2);
  cvt_bf16<<<512, 256, 0, stream>>>(Wout_f, wout_b, (long)EE * KOUT);
  build_xs<<<2048, 256, 0, stream>>>(gt, ctx_f, xs_b);

  igates_gemm<<<dim3(32, 128), 256, 0, stream>>>(xs_b, wih_b, bih, bhh, igates);
  lstm_recur<<<64, 256, 0, stream>>>(igates, whh_b, h_all, bar);
  attn_energy<<<BB * 32, 256, 0, stream>>>(h_all, ctx_b, energy);
  softmax_rows<<<BB * LL, 256, 0, stream>>>(energy, score);
  attn_ctx<<<BB * 32, 256, 0, stream>>>(score, ctx_b, ctx_all);
  out_logits<<<128, 256, 0, stream>>>(h_all, ctx_all, wout_b, bout, logits);
  out_logsoftmax<<<MX, 128, 0, stream>>>(logits, out);
}
